// Llama4TextMoe_6863357739472
// MI455X (gfx1250) — compile-verified, matrix operand order checked by reading, not run on hardware
//
#include <hip/hip_runtime.h>

#define T_TOK 2048
#define HDIM  2048
#define FDIM  4096
#define NEXP  8
#define NT_MAX (T_TOK/128 + NEXP)   // worst-case routed tiles (24)

typedef __attribute__((ext_vector_type(16))) __bf16 v16bf;
typedef __attribute__((ext_vector_type(8)))  float  v8f;

union FragU { uint4 q[2]; v16bf v; };

// round-half-up bf16 (<=1ulp from RNE; bf16 noise level)
__device__ __forceinline__ unsigned short f32_bf16(float f) {
    return (unsigned short)((__float_as_uint(f) + 0x8000u) >> 16);
}

// two floats -> packed bf16 pair: 2x v_add + 1x v_perm_b32
__device__ __forceinline__ unsigned int pack_bf16(float lo, float hi) {
    unsigned int ulo = __float_as_uint(lo) + 0x8000u;
    unsigned int uhi = __float_as_uint(hi) + 0x8000u;
    return __builtin_amdgcn_perm(uhi, ulo, 0x07060302u);  // {uhi.b3,uhi.b2,ulo.b3,ulo.b2}
}

__device__ __forceinline__ float fast_sigmoid(float x) {
    return __builtin_amdgcn_rcpf(1.f + __expf(-x));      // v_rcp_f32, no IEEE div
}

// Load one 16x32 (A) or 32x16-transposed (B) bf16 fragment from an LDS tile
// stored with 32 contiguous K-elements per row/col.  Per ISA layout:
// lane&15 selects row/col, lane>>4 selects the K-half (8 elems + 8 elems@+16).
__device__ __forceinline__ v16bf frag_ld(const unsigned short* tile, int idx, int lane) {
    const unsigned short* p = tile + idx * 32 + ((lane >> 4) << 3);
    FragU fu;
    fu.q[0] = *(const uint4*)(p);
    fu.q[1] = *(const uint4*)(p + 16);
    return fu.v;
}

// ----------------------------------------------- Tensor Data Mover (TDM) ----
#if __has_builtin(__builtin_amdgcn_tensor_load_to_lds)
#define USE_TDM 1
typedef unsigned int u32x4 __attribute__((ext_vector_type(4)));
typedef int          i32x8 __attribute__((ext_vector_type(8)));
typedef int          i32x4 __attribute__((ext_vector_type(4)));

// DMA a 2D bf16 tile [rows(<=128) x 32 elems] (row stride = stride elems)
// from global into LDS (row-major, 64B per row).  tensor_dim1 = rows, so
// tile rows beyond `rows` are OOB and fill with zeros (discarded later).
__device__ __forceinline__ void tdm_load_a_tile(unsigned lds_addr, const void* gptr,
                                                int rows, int stride_elems) {
    unsigned long long ga = (unsigned long long)(size_t)gptr;
    u32x4 g0;
    g0[0] = 1u;                                              // count=1, user desc
    g0[1] = lds_addr;                                        // LDS byte address
    g0[2] = (unsigned)ga;                                    // global_addr[31:0]
    g0[3] = (unsigned)((ga >> 32) & 0x1FFFFFFu) | (2u << 30); // addr[56:32] | type=2
    i32x8 g1;
    g1[0] = (1 << 16);           // workgroup_mask=0 | data_size=1 (2 bytes)
    g1[1] = (32 << 16);          // tensor_dim0[15:0]=32 in bits[31:16]
    g1[2] = (rows << 16);        // tensor_dim0[31:16]=0 | tensor_dim1[15:0]=rows
    g1[3] = (32 << 16);          // tensor_dim1[31:16]=0 | tile_dim0=32
    g1[4] = 128;                 // tile_dim1=128 | tile_dim2=0
    g1[5] = stride_elems;        // tensor_dim0_stride[31:0]
    g1[6] = 0;                   // stride hi | tensor_dim1_stride lo (unused, 2D)
    g1[7] = 0;
    // 6-arg toolchain form: groups 2/3 disabled (2D tensor), trailing group zero
    __builtin_amdgcn_tensor_load_to_lds(g0, g1, (i32x4)0, (i32x4)0, (i32x8)0, 0);
}
#else
#define USE_TDM 0
#endif

// ---------------------------------------------------------------- router ----
__global__ __launch_bounds__(256)
void router_kernel(const float* __restrict__ h, const float* __restrict__ rw,
                   float* __restrict__ logits, int* __restrict__ tok_expert,
                   float* __restrict__ tok_score) {
    int lane = threadIdx.x & 31;
    int t = blockIdx.x * 8 + (threadIdx.x >> 5);          // one wave per token
    float acc[NEXP];
#pragma unroll
    for (int e = 0; e < NEXP; ++e) acc[e] = 0.f;
    const float* hr = h + (size_t)t * HDIM;
    for (int k = lane; k < HDIM; k += 32) {
        float hv = hr[k];
        const float* wr = rw + (size_t)k * NEXP;
#pragma unroll
        for (int e = 0; e < NEXP; ++e) acc[e] = fmaf(hv, wr[e], acc[e]);
    }
#pragma unroll
    for (int e = 0; e < NEXP; ++e)
#pragma unroll
        for (int off = 16; off; off >>= 1) acc[e] += __shfl_xor(acc[e], off, 32);
    if (lane == 0) {
        float best = acc[0]; int bi = 0;
#pragma unroll
        for (int e = 1; e < NEXP; ++e) if (acc[e] > best) { best = acc[e]; bi = e; }
        tok_expert[t] = bi;
        tok_score[t]  = fast_sigmoid(best);
#pragma unroll
        for (int e = 0; e < NEXP; ++e) logits[(size_t)t * NEXP + e] = acc[e];
    }
}

// ------------------------------------------------------------------ scan ----
__global__ void scan_kernel(const int* __restrict__ te, int* __restrict__ counters,
                            int* __restrict__ tile_e, int* __restrict__ tile_s,
                            int* __restrict__ tile_c, int* __restrict__ n_tiles) {
    __shared__ int cnt[NEXP];
    if (threadIdx.x < NEXP) cnt[threadIdx.x] = 0;
    __syncthreads();
    for (int t = threadIdx.x; t < T_TOK; t += 256) atomicAdd(&cnt[te[t]], 1);
    __syncthreads();
    if (threadIdx.x == 0) {
        int off = 0, nt = 0;
        for (int e = 0; e < NEXP; ++e) {
            counters[e] = off;                       // scatter write cursor
            int c = cnt[e];
            for (int s = 0; s < c; s += 128) {
                tile_e[nt] = e; tile_s[nt] = off + s;
                tile_c[nt] = (c - s < 128) ? (c - s) : 128; ++nt;
            }
            off += c;
        }
        *n_tiles = nt;
    }
}

__global__ void scatter_kernel(const int* __restrict__ te, int* __restrict__ counters,
                               int* __restrict__ perm) {
    int t = blockIdx.x * 256 + threadIdx.x;
    int slot = atomicAdd(&counters[te[t]], 1);
    perm[slot] = t;
}

// --------------------------------------------------- gate/up + SiLU GEMM ----
// C[128 x (64 gate | 64 up)] per block; act = silu(gate)*up -> bf16.
// Double-buffered LDS; global pointers advance by the K-step.
template <bool ROUTED>
__global__ __launch_bounds__(256)
void gemm_gateup(const float* __restrict__ h, const int* __restrict__ perm,
                 const float* __restrict__ score, const int* __restrict__ tile_e,
                 const int* __restrict__ tile_s, const int* __restrict__ tile_c,
                 const int* __restrict__ n_tiles, const float* __restrict__ Wg_base,
                 const float* __restrict__ Wu_base, unsigned short* __restrict__ act) {
    __shared__ __align__(16) unsigned short As[2][128 * 32];   // [row][k]
    __shared__ __align__(16) unsigned short Bg[2][64 * 32];    // [n][k]
    __shared__ __align__(16) unsigned short Bu[2][64 * 32];

    int tile = blockIdx.y;
    int e = 0, row0, rcnt = 128;
    if constexpr (ROUTED) {
        if (tile >= *n_tiles) return;
        e = tile_e[tile]; row0 = tile_s[tile]; rcnt = tile_c[tile];
    } else {
        row0 = tile * 128;
    }
    const float *Wg, *Wu; int wld;
    if constexpr (ROUTED) {
        const float* base = Wg_base + (size_t)e * HDIM * (2 * FDIM);
        Wg = base; Wu = base + FDIM; wld = 2 * FDIM;
    } else { Wg = Wg_base; Wu = Wu_base; wld = FDIM; }

    const int n0 = blockIdx.x * 64;
    const int tid = threadIdx.x, lane = tid & 31, wave = tid >> 5;
    const int rg = wave >> 1, cg = wave & 1;

    // A-tile loader: 2 threads per row, 16 k each
    const int arow = tid >> 1, akofs = (tid & 1) * 16;
    float ascale = 1.f;
    const float* aP;
    {
        int r = (arow < rcnt) ? arow : 0;
        int tok = ROUTED ? perm[row0 + r] : (row0 + arow);
        aP = h + (size_t)tok * HDIM + akofs;
        if (ROUTED) ascale = score[tok];
    }
    // B-tile loader: one k-PAIR x 4 n per thread -> packed dword LDS stores
    const int kp = tid >> 4;            // 0..15 (k pair)
    const int n4 = (tid & 15) * 4;      // 0..60
    const float* gP = Wg + (size_t)(2 * kp) * wld + n0 + n4;
    const float* uP = Wu + (size_t)(2 * kp) * wld + n0 + n4;
    const size_t wstep = (size_t)32 * wld;

    v8f accg[2][2], accu[2][2];
#pragma unroll
    for (int i = 0; i < 2; ++i)
#pragma unroll
        for (int j = 0; j < 2; ++j) { accg[i][j] = {}; accu[i][j] = {}; }

    union AV { float4 v4[4]; float f[16]; } aU;
    union BV { float4 v4[2]; float f[8];  } gU, uU;

    const int NK = HDIM / 32;
#define GU_LOAD()                                                                \
    {   const float4* ap = (const float4*)aP;                                    \
        aU.v4[0] = ap[0]; aU.v4[1] = ap[1]; aU.v4[2] = ap[2]; aU.v4[3] = ap[3];  \
        gU.v4[0] = *(const float4*)gP; gU.v4[1] = *(const float4*)(gP + wld);    \
        uU.v4[0] = *(const float4*)uP; uU.v4[1] = *(const float4*)(uP + wld);    \
        aP += 32; gP += wstep; uP += wstep; }
#define GU_STORE(P)                                                              \
    {   unsigned int apk[8];                                                     \
        _Pragma("unroll")                                                        \
        for (int q = 0; q < 8; ++q)                                              \
            apk[q] = pack_bf16(aU.f[2*q] * ascale, aU.f[2*q+1] * ascale);        \
        uint4* asd = (uint4*)&As[P][arow * 32 + akofs];                          \
        asd[0] = *(uint4*)&apk[0]; asd[1] = *(uint4*)&apk[4];                    \
        _Pragma("unroll")                                                        \
        for (int j = 0; j < 4; ++j) {                                            \
            *(unsigned int*)&Bg[P][(n4 + j) * 32 + 2 * kp] =                     \
                pack_bf16(gU.f[j], gU.f[4 + j]);                                 \
            *(unsigned int*)&Bu[P][(n4 + j) * 32 + 2 * kp] =                     \
                pack_bf16(uU.f[j], uU.f[4 + j]);                                 \
        }  }

    GU_LOAD()
    GU_STORE(0)
    for (int it = 0; it < NK; ++it) {
        const int p = it & 1;
        __syncthreads();
        if (it + 1 < NK) GU_LOAD()

        v16bf a0 = frag_ld(As[p], rg * 32 +  0 + (lane & 15), lane);
        v16bf a1 = frag_ld(As[p], rg * 32 + 16 + (lane & 15), lane);
        v16bf g0 = frag_ld(Bg[p], cg * 32 +  0 + (lane & 15), lane);
        v16bf g1 = frag_ld(Bg[p], cg * 32 + 16 + (lane & 15), lane);
        v16bf u0 = frag_ld(Bu[p], cg * 32 +  0 + (lane & 15), lane);
        v16bf u1 = frag_ld(Bu[p], cg * 32 + 16 + (lane & 15), lane);

        accg[0][0] = __builtin_amdgcn_wmma_f32_16x16x32_bf16(false, a0, false, g0, (short)0, accg[0][0], false, false);
        accg[0][1] = __builtin_amdgcn_wmma_f32_16x16x32_bf16(false, a0, false, g1, (short)0, accg[0][1], false, false);
        accg[1][0] = __builtin_amdgcn_wmma_f32_16x16x32_bf16(false, a1, false, g0, (short)0, accg[1][0], false, false);
        accg[1][1] = __builtin_amdgcn_wmma_f32_16x16x32_bf16(false, a1, false, g1, (short)0, accg[1][1], false, false);
        accu[0][0] = __builtin_amdgcn_wmma_f32_16x16x32_bf16(false, a0, false, u0, (short)0, accu[0][0], false, false);
        accu[0][1] = __builtin_amdgcn_wmma_f32_16x16x32_bf16(false, a0, false, u1, (short)0, accu[0][1], false, false);
        accu[1][0] = __builtin_amdgcn_wmma_f32_16x16x32_bf16(false, a1, false, u0, (short)0, accu[1][0], false, false);
        accu[1][1] = __builtin_amdgcn_wmma_f32_16x16x32_bf16(false, a1, false, u1, (short)0, accu[1][1], false, false);

        if (it + 1 < NK) GU_STORE(p ^ 1)
    }
#undef GU_LOAD
#undef GU_STORE

    // epilogue: act = silu(gate)*up -> bf16, permuted-slot-major [slot][F]
#pragma unroll
    for (int rt = 0; rt < 2; ++rt)
#pragma unroll
        for (int ct = 0; ct < 2; ++ct)
#pragma unroll
            for (int i = 0; i < 8; ++i) {
                int r = rg * 32 + rt * 16 + ((lane >> 4) << 3) + i;
                if (r < rcnt) {
                    int n = n0 + cg * 32 + ct * 16 + (lane & 15);
                    float g = accg[rt][ct][i], u = accu[rt][ct][i];
                    act[(size_t)(row0 + r) * FDIM + n] =
                        f32_bf16(u * g * fast_sigmoid(g));
                }
            }
}

// -------------------------------------------------------------- down GEMM ----
// A tile (bf16, contiguous slot rows) is DMA'd by the Tensor Data Mover when
// available; B tile is converted f32->bf16 through VGPRs.
template <bool ROUTED>
__global__ __launch_bounds__(256)
void gemm_down(const unsigned short* __restrict__ act, const int* __restrict__ perm,
               const int* __restrict__ tile_e, const int* __restrict__ tile_s,
               const int* __restrict__ tile_c, const int* __restrict__ n_tiles,
               const float* __restrict__ W_base, float* __restrict__ out) {
    __shared__ __align__(16) unsigned short As[2][128 * 32];   // [row][k]
    __shared__ __align__(16) unsigned short Bs[2][128 * 32];   // [n][k]

    int tile = blockIdx.y;
    int e = 0, row0, rcnt = 128;
    if constexpr (ROUTED) {
        if (tile >= *n_tiles) return;
        e = tile_e[tile]; row0 = tile_s[tile]; rcnt = tile_c[tile];
    } else {
        row0 = tile * 128;
    }
    const float* W = ROUTED ? (W_base + (size_t)e * FDIM * HDIM) : W_base;  // [F][H]

    const int n0 = blockIdx.x * 128;
    const int tid = threadIdx.x, lane = tid & 31, wave = tid >> 5;
    const int rg = wave >> 1, cg = wave & 1;

    // B loader: one k-pair x 8 n per thread
    const int kp = tid >> 4;            // 0..15
    const int n8 = (tid & 15) * 8;      // 0..120
    const float* bP = W + (size_t)(2 * kp) * HDIM + n0 + n8;

#if USE_TDM
    const unsigned short* aTile = act + (size_t)row0 * FDIM;   // contiguous slot rows
    const unsigned ldsA0 = (unsigned)(size_t)&As[0][0];        // LDS byte offset
    const unsigned ldsA1 = (unsigned)(size_t)&As[1][0];
#else
    const int arow = tid >> 1, akofs = (tid & 1) * 16;
    const int aslot = row0 + ((arow < rcnt) ? arow : 0);
    const unsigned short* aP = act + (size_t)aslot * FDIM + akofs;
    uint4 aQ[2];
#endif

    v8f acc[2][4];
#pragma unroll
    for (int i = 0; i < 2; ++i)
#pragma unroll
        for (int j = 0; j < 4; ++j) acc[i][j] = {};

    union BV { float4 v4[4]; float f[16]; } bU;   // f[0..7]=row k, f[8..15]=row k+1

    const int NK = FDIM / 32;
#define DN_LOADB()                                                               \
    {   bU.v4[0] = *(const float4*)bP;          bU.v4[1] = *(const float4*)(bP + 4); \
        bU.v4[2] = *(const float4*)(bP + HDIM); bU.v4[3] = *(const float4*)(bP + HDIM + 4); \
        bP += (size_t)32 * HDIM; }
#define DN_STOREB(P)                                                             \
    {   _Pragma("unroll")                                                        \
        for (int j = 0; j < 8; ++j)                                              \
            *(unsigned int*)&Bs[P][(n8 + j) * 32 + 2 * kp] =                     \
                pack_bf16(bU.f[j], bU.f[8 + j]);  }
#if USE_TDM
#define DN_LOADA(IT)  { if (wave == 0) \
        tdm_load_a_tile((((IT) & 1) ? ldsA1 : ldsA0), aTile + (IT) * 32, rcnt, FDIM); }
#define DN_STOREA(P)  {}
#define DN_WAITA()    { if (wave == 0) __builtin_amdgcn_s_wait_tensorcnt(0); }
#else
#define DN_LOADA(IT)  { const uint4* ap = (const uint4*)aP; aQ[0] = ap[0]; aQ[1] = ap[1]; aP += 32; }
#define DN_STOREA(P)  { uint4* asd = (uint4*)&As[P][arow * 32 + akofs]; \
                        asd[0] = aQ[0]; asd[1] = aQ[1]; }
#define DN_WAITA()    {}
#endif

    DN_LOADA(0)
    DN_LOADB()
    DN_STOREA(0)
    DN_STOREB(0)
    for (int it = 0; it < NK; ++it) {
        const int p = it & 1;
        DN_WAITA()            // wave 0: TENSORcnt==0 for buffer p, then publish
        __syncthreads();
        if (it + 1 < NK) {
            DN_LOADA(it + 1)  // TDM: DMA next A tile into buffer p^1
            DN_LOADB()
        }

        v16bf a0 = frag_ld(As[p], rg * 32 +  0 + (lane & 15), lane);
        v16bf a1 = frag_ld(As[p], rg * 32 + 16 + (lane & 15), lane);
        v16bf b0 = frag_ld(Bs[p], cg * 64 +  0 + (lane & 15), lane);
        v16bf b1 = frag_ld(Bs[p], cg * 64 + 16 + (lane & 15), lane);
        v16bf b2 = frag_ld(Bs[p], cg * 64 + 32 + (lane & 15), lane);
        v16bf b3 = frag_ld(Bs[p], cg * 64 + 48 + (lane & 15), lane);

        acc[0][0] = __builtin_amdgcn_wmma_f32_16x16x32_bf16(false, a0, false, b0, (short)0, acc[0][0], false, false);
        acc[0][1] = __builtin_amdgcn_wmma_f32_16x16x32_bf16(false, a0, false, b1, (short)0, acc[0][1], false, false);
        acc[0][2] = __builtin_amdgcn_wmma_f32_16x16x32_bf16(false, a0, false, b2, (short)0, acc[0][2], false, false);
        acc[0][3] = __builtin_amdgcn_wmma_f32_16x16x32_bf16(false, a0, false, b3, (short)0, acc[0][3], false, false);
        acc[1][0] = __builtin_amdgcn_wmma_f32_16x16x32_bf16(false, a1, false, b0, (short)0, acc[1][0], false, false);
        acc[1][1] = __builtin_amdgcn_wmma_f32_16x16x32_bf16(false, a1, false, b1, (short)0, acc[1][1], false, false);
        acc[1][2] = __builtin_amdgcn_wmma_f32_16x16x32_bf16(false, a1, false, b2, (short)0, acc[1][2], false, false);
        acc[1][3] = __builtin_amdgcn_wmma_f32_16x16x32_bf16(false, a1, false, b3, (short)0, acc[1][3], false, false);

        if (it + 1 < NK) { DN_STOREA(p ^ 1) DN_STOREB(p ^ 1) }
    }
#undef DN_LOADA
#undef DN_LOADB
#undef DN_STOREA
#undef DN_STOREB
#undef DN_WAITA

#pragma unroll
    for (int rt = 0; rt < 2; ++rt)
#pragma unroll
        for (int ct = 0; ct < 4; ++ct)
#pragma unroll
            for (int i = 0; i < 8; ++i) {
                int r = rg * 32 + rt * 16 + ((lane >> 4) << 3) + i;
                if (r < rcnt) {
                    int c = n0 + cg * 64 + ct * 16 + (lane & 15);
                    float v = acc[rt][ct][i];
                    if constexpr (ROUTED) {
                        int orow = perm[row0 + r];
                        out[(size_t)orow * HDIM + c] += v;   // after shared wrote '='
                    } else {
                        out[(size_t)(row0 + r) * HDIM + c] = v;
                    }
                }
            }
}

// ---------------------------------------------------------------- launch ----
extern "C" void kernel_launch(void* const* d_in, const int* in_sizes, int n_in,
                              void* d_out, int out_size, void* d_ws, size_t ws_size,
                              hipStream_t stream) {
    const float* h   = (const float*)d_in[0];
    const float* rw  = (const float*)d_in[1];
    const float* gup = (const float*)d_in[2];
    const float* dwn = (const float*)d_in[3];
    const float* sg  = (const float*)d_in[4];
    const float* su  = (const float*)d_in[5];
    const float* sd  = (const float*)d_in[6];

    float* out    = (float*)d_out;                       // [T,H]
    float* logits = out + (size_t)T_TOK * HDIM;          // [T,E]

    char* ws = (char*)d_ws;
    size_t off = 0;
    auto take = [&](size_t bytes) { size_t c = off; off = (off + bytes + 255) & ~(size_t)255; return c; };
    int*   tok_expert = (int*)  (ws + take(sizeof(int)   * T_TOK));
    float* tok_score  = (float*)(ws + take(sizeof(float) * T_TOK));
    int*   perm       = (int*)  (ws + take(sizeof(int)   * T_TOK));
    int*   counters   = (int*)  (ws + take(sizeof(int)   * NEXP));
    int*   tile_e     = (int*)  (ws + take(sizeof(int)   * NT_MAX));
    int*   tile_s     = (int*)  (ws + take(sizeof(int)   * NT_MAX));
    int*   tile_c     = (int*)  (ws + take(sizeof(int)   * NT_MAX));
    int*   n_tiles    = (int*)  (ws + take(sizeof(int)));
    unsigned short* act_r = (unsigned short*)(ws + take((size_t)T_TOK * FDIM * 2));
    unsigned short* act_s = (unsigned short*)(ws + take((size_t)T_TOK * FDIM * 2));
    (void)in_sizes; (void)n_in; (void)out_size; (void)ws_size;

    router_kernel<<<T_TOK / 8, 256, 0, stream>>>(h, rw, logits, tok_expert, tok_score);
    scan_kernel<<<1, 256, 0, stream>>>(tok_expert, counters, tile_e, tile_s, tile_c, n_tiles);
    scatter_kernel<<<T_TOK / 256, 256, 0, stream>>>(tok_expert, counters, perm);

    // shared expert: gate/up + silu -> act_s ; down -> out (plain store)
    gemm_gateup<false><<<dim3(FDIM / 64, T_TOK / 128), 256, 0, stream>>>(
        h, nullptr, nullptr, nullptr, nullptr, nullptr, nullptr, sg, su, act_s);
    gemm_down<false><<<dim3(HDIM / 128, T_TOK / 128), 256, 0, stream>>>(
        act_s, nullptr, nullptr, nullptr, nullptr, nullptr, sd, out);

    // routed experts (top-1, grouped by permutation): act_r ; down accumulates
    gemm_gateup<true><<<dim3(FDIM / 64, NT_MAX), 256, 0, stream>>>(
        h, perm, tok_score, tile_e, tile_s, tile_c, n_tiles, gup, gup, act_r);
    gemm_down<true><<<dim3(HDIM / 128, NT_MAX), 256, 0, stream>>>(
        act_r, perm, tile_e, tile_s, tile_c, n_tiles, dwn, out);
}
